// Encoder_49563922596698
// MI455X (gfx1250) — matrix operand order repaired and checked by reading, not compiled, not run on hardware
//
#include <hip/hip_runtime.h>
#include <hip/hip_bf16.h>

// ---------------------------------------------------------------------------
// Conv-transformer encoder for MI455X (gfx1250), bf16 WMMA everywhere.
// B=32 T=1024 D=256 H=2 dk=128 D_HID=1024 L=4, conv1 k=9 pad=4, conv2 k=1.
// Round 3: fix async-to-LDS builtin pointee type (int4 GCC vector in AS1/AS3).
// ---------------------------------------------------------------------------

typedef __bf16 bf16_t;
typedef __attribute__((ext_vector_type(16))) __bf16 v16bf;
typedef __attribute__((ext_vector_type(8)))  float  v8f;

union FragBF { uint4 q[2]; v16bf v; };

__device__ __forceinline__ unsigned short f2bf(float f) {
  unsigned int u = __float_as_uint(f);
  u += 0x7FFFu + ((u >> 16) & 1u);          // round to nearest even
  return (unsigned short)(u >> 16);
}

#if __has_builtin(__builtin_amdgcn_global_load_async_to_lds_b128)
#define HAVE_ASYNC_LDS 1
#pragma message("gfx1250: global_load_async_to_lds_b128 builtin AVAILABLE")
#else
#define HAVE_ASYNC_LDS 0
#pragma message("gfx1250: global_load_async_to_lds_b128 builtin NOT available")
#endif

// Builtin prototype wants: (int4 __device__ *, int4 __shared__ *, Imm, Imm)
typedef int v4i_gcc __attribute__((vector_size(16)));
typedef __attribute__((address_space(1))) v4i_gcc* gv4i_p;
typedef __attribute__((address_space(3))) v4i_gcc* lv4i_p;

// 16B global -> LDS copy: async (direct-to-LDS, ASYNCcnt) when available.
__device__ __forceinline__ void copy16_g2l(const void* g, void* l) {
#if HAVE_ASYNC_LDS
  __builtin_amdgcn_global_load_async_to_lds_b128((gv4i_p)(g), (lv4i_p)(l), 0, 0);
#else
  *reinterpret_cast<uint4*>(l) = *reinterpret_cast<const uint4*>(g);
#endif
}

__device__ __forceinline__ void async_wait() {
#if HAVE_ASYNC_LDS
#if __has_builtin(__builtin_amdgcn_s_wait_asynccnt)
  __builtin_amdgcn_s_wait_asynccnt(0);
#else
  asm volatile("s_wait_asynccnt 0x0" ::: "memory");
#endif
#endif
}

#define NB   32
#define NT   1024
#define ND   256
#define NBT  32768        // NB*NT
#define NHID 1024
#define NLAY 4

// ---------------------------------------------------------------------------
// Embedding + non_pad
// ---------------------------------------------------------------------------
__global__ __launch_bounds__(256) void embed_k(
    const int* __restrict__ seq, const int* __restrict__ pos,
    const float* __restrict__ wemb, const float* __restrict__ pemb,
    float* __restrict__ x, float* __restrict__ non_pad) {
  int row = blockIdx.x;
  int d   = threadIdx.x;
  int s = seq[row];
  int p = pos[row];
  x[(size_t)row * ND + d] = wemb[(size_t)s * ND + d] + pemb[(size_t)p * ND + d];
  if (d == 0) non_pad[row] = (s != 0) ? 1.0f : 0.0f;
}

// ---------------------------------------------------------------------------
// Weight conversion f32 -> bf16 (flat)
// ---------------------------------------------------------------------------
__global__ __launch_bounds__(256) void cvt_k(
    const float* __restrict__ src, unsigned short* __restrict__ dst, int n) {
  int i = blockIdx.x * 256 + threadIdx.x;
  if (i < n) dst[i] = f2bf(src[i]);
}

// conv1_w [L,1024,256,9] -> [L,1024,9,256] bf16 (K = tap*256 + d contiguous)
__global__ __launch_bounds__(256) void cvt_conv1_k(
    const float* __restrict__ src, unsigned short* __restrict__ dst, int n) {
  int i = blockIdx.x * 256 + threadIdx.x;
  if (i >= n) return;
  int d   = i & 255;
  int r   = i >> 8;
  int tap = r % 9;
  int lo  = r / 9;                       // l*1024 + o
  dst[i] = f2bf(src[((size_t)lo * 256 + d) * 9 + tap]);
}

// ---------------------------------------------------------------------------
// LayerNorm over D=256, one wave per row, bf16 out
// ---------------------------------------------------------------------------
__global__ __launch_bounds__(256) void ln_k(
    const float* __restrict__ x, const float* __restrict__ g,
    const float* __restrict__ bta, unsigned short* __restrict__ h) {
  int lane = threadIdx.x & 31;
  int wv   = threadIdx.x >> 5;
  int row  = blockIdx.x * 8 + wv;
  const float* xr = x + (size_t)row * ND;
  float v[8];
  float s = 0.f;
#pragma unroll
  for (int i = 0; i < 8; ++i) { v[i] = xr[i * 32 + lane]; s += v[i]; }
#pragma unroll
  for (int off = 16; off > 0; off >>= 1) s += __shfl_xor(s, off);
  float mean = s * (1.0f / 256.0f);
  float q = 0.f;
#pragma unroll
  for (int i = 0; i < 8; ++i) { float d = v[i] - mean; q += d * d; }
#pragma unroll
  for (int off = 16; off > 0; off >>= 1) q += __shfl_xor(q, off);
  float rinv = rsqrtf(q * (1.0f / 256.0f) + 1e-5f);
  unsigned short* hr = h + (size_t)row * ND;
#pragma unroll
  for (int i = 0; i < 8; ++i) {
    int d = i * 32 + lane;
    hr[d] = f2bf((v[i] - mean) * rinv * g[d] + bta[d]);
  }
}

// ---------------------------------------------------------------------------
// Generic bf16 WMMA GEMM: C[M,N] = A[M,K] @ B[N,K]^T (+bias, modes)
//   mode 0: bf16 out          mode 1: bf16 out + relu
//   mode 2: f32 out, += resid, *= mask[row]   (residual/mask fused epilogue)
// IM2COL: A is h[B,T,D] bf16; logical K = tap*256 + d (conv k=9 pad=4).
// WG: 256 thr = 8 waves, tile BM=128 BN=128 BK=32; wave tile 32x64 (2x4
// frags -> 8 wmma per k-step, 12 ds_load_b128 per k-step per wave).
// ---------------------------------------------------------------------------
template <int IM2COL>
__global__ __launch_bounds__(256) void gemm_k(
    const unsigned short* __restrict__ A, int lda,
    const unsigned short* __restrict__ Bw, int ldb,
    const float* __restrict__ bias,
    float* __restrict__ outF, unsigned short* __restrict__ outB, int ldc,
    const float* __restrict__ resid, const float* __restrict__ mask,
    int K, int mode) {
  __shared__ __align__(16) unsigned short As[128][40];  // 40*2B = 80B rows
  __shared__ __align__(16) unsigned short Bs[128][40];

  const int tid   = threadIdx.x;
  const int mbase = blockIdx.y * 128;
  const int nbase = blockIdx.x * 128;
  const int lane  = tid & 31;
  const int wv    = tid >> 5;
  const int wm    = (wv & 3) * 32;     // 4 wave-rows * 32
  const int wn    = (wv >> 2) * 64;    // 2 wave-cols * 64
  const int lrow  = lane & 15;
  const int lhi   = lane >> 4;

  v8f zero = {0, 0, 0, 0, 0, 0, 0, 0};
  v8f acc[2][4];
#pragma unroll
  for (int i = 0; i < 2; ++i)
#pragma unroll
    for (int j = 0; j < 4; ++j) acc[i][j] = zero;

  const int nKt = K >> 5;
  for (int kt = 0; kt < nKt; ++kt) {
    const int k0 = kt << 5;
    __syncthreads();
    // ---- stage A tile 128x32 (two 16B units per thread) ----
#pragma unroll
    for (int uu = 0; uu < 2; ++uu) {
      int u = tid + uu * 256;
      int r = u >> 2, cu = u & 3;
      if (!IM2COL) {
        copy16_g2l(A + (size_t)(mbase + r) * lda + k0 + cu * 8, &As[r][cu * 8]);
      } else {
        int tap = k0 >> 8;                  // 32-wide K tile stays in one tap
        int d0  = (k0 & 255) + cu * 8;
        int m = mbase + r;
        int bb = m >> 10, t = m & 1023;
        int st = t + tap - 4;
        uint4 val;
        if (st >= 0 && st < NT)
          val = *reinterpret_cast<const uint4*>(
              A + ((size_t)((bb << 10) + st) * ND + d0));
        else
          val = make_uint4(0, 0, 0, 0);
        *reinterpret_cast<uint4*>(&As[r][cu * 8]) = val;
      }
    }
    // ---- stage B tile 128x32 (two 16B units per thread) ----
#pragma unroll
    for (int uu = 0; uu < 2; ++uu) {
      int u = tid + uu * 256;
      int r = u >> 2, cu = u & 3;
      copy16_g2l(Bw + (size_t)(nbase + r) * ldb + k0 + cu * 8, &Bs[r][cu * 8]);
    }
    async_wait();
    __syncthreads();
    // ---- fragments (ISA 16-bit A 16x32 / B 32x16 lane layouts) ----
    FragBF a[2], bb[4];
#pragma unroll
    for (int fm = 0; fm < 2; ++fm) {
      int rr = wm + fm * 16 + lrow;
      int kb = lhi * 8;
      a[fm].q[0] = *reinterpret_cast<const uint4*>(&As[rr][kb]);
      a[fm].q[1] = *reinterpret_cast<const uint4*>(&As[rr][kb + 16]);
    }
#pragma unroll
    for (int fn = 0; fn < 4; ++fn) {
      int rr = wn + fn * 16 + lrow;
      int kb = lhi * 8;
      bb[fn].q[0] = *reinterpret_cast<const uint4*>(&Bs[rr][kb]);
      bb[fn].q[1] = *reinterpret_cast<const uint4*>(&Bs[rr][kb + 16]);
    }
#pragma unroll
    for (int fm = 0; fm < 2; ++fm)
#pragma unroll
      for (int fn = 0; fn < 4; ++fn)
        acc[fm][fn] = __builtin_amdgcn_wmma_f32_16x16x32_bf16(
            false, a[fm].v, false, bb[fn].v, (short)0, acc[fm][fn], false, false);
  }
  // ---- epilogue ----
#pragma unroll
  for (int fm = 0; fm < 2; ++fm)
#pragma unroll
    for (int fn = 0; fn < 4; ++fn)
#pragma unroll
      for (int r = 0; r < 8; ++r) {
        int gR = mbase + wm + fm * 16 + r + lhi * 8;
        int gC = nbase + wn + fn * 16 + lrow;
        float v = acc[fm][fn][r] + bias[gC];
        size_t off = (size_t)gR * ldc + gC;
        if (mode == 0) {
          outB[off] = f2bf(v);
        } else if (mode == 1) {
          outB[off] = f2bf(v > 0.f ? v : 0.f);
        } else {
          outF[off] = (v + resid[off]) * mask[gR];
        }
      }
}

// ---------------------------------------------------------------------------
// Flash attention: WG = (qtile, head, batch), 128 thr = 4 waves,
// each wave owns 16 q rows. dk=128, key tiles of 64 (16 tiles).
// qkv rows: [q(256)|k(256)|v(256)], head h at +h*128 within each.
// ---------------------------------------------------------------------------
__global__ __launch_bounds__(128) void attn_k(
    const unsigned short* __restrict__ qkv,
    const int* __restrict__ seq,
    unsigned short* __restrict__ o) {
  __shared__ __align__(16) unsigned short Qs[64][136];
  __shared__ __align__(16) unsigned short Ks[64][136];
  __shared__ __align__(16) unsigned short Vt[128][72];   // transposed [dk][key]
  __shared__ __align__(16) unsigned short Ps[64][72];

  const int tid  = threadIdx.x;
  const int lane = tid & 31;
  const int wv   = tid >> 5;
  const int lrow = lane & 15, lhi = lane >> 4;
  const int qt = blockIdx.x, hd = blockIdx.y, b = blockIdx.z;
  const int qbase = qt * 64;
  const size_t rs = 768;
  const unsigned short* qptr = qkv + (size_t)(b * NT + qbase) * rs + hd * 128;

  // load Q tile 64x128 (async path)
#pragma unroll
  for (int j = 0; j < 8; ++j) {
    int u = tid + j * 128;
    int r = u >> 4, cu = u & 15;
    copy16_g2l(qptr + (size_t)r * rs + cu * 8, &Qs[r][cu * 8]);
  }
  async_wait();
  __syncthreads();

  FragBF aQ[4];
  const int qr = wv * 16;
#pragma unroll
  for (int ks = 0; ks < 4; ++ks) {
    int rr = qr + lrow;
    int kb = ks * 32 + lhi * 8;
    aQ[ks].q[0] = *reinterpret_cast<const uint4*>(&Qs[rr][kb]);
    aQ[ks].q[1] = *reinterpret_cast<const uint4*>(&Qs[rr][kb + 16]);
  }

  v8f zero = {0, 0, 0, 0, 0, 0, 0, 0};
  float mrow[8], lsum[8];
  v8f Oacc[8];
#pragma unroll
  for (int r = 0; r < 8; ++r) { mrow[r] = -1e30f; lsum[r] = 0.f; }
#pragma unroll
  for (int fn = 0; fn < 8; ++fn) Oacc[fn] = zero;

  for (int kt = 0; kt < 16; ++kt) {
    __syncthreads();
    const unsigned short* kptr = qkv + (size_t)(b * NT + kt * 64) * rs + 256 + hd * 128;
    const unsigned short* vptr = qkv + (size_t)(b * NT + kt * 64) * rs + 512 + hd * 128;
#pragma unroll
    for (int j = 0; j < 8; ++j) {
      int u = tid + j * 128;
      int r = u >> 4, cu = u & 15;
      copy16_g2l(kptr + (size_t)r * rs + cu * 8, &Ks[r][cu * 8]);    // async
      uint4 vvv = *reinterpret_cast<const uint4*>(vptr + (size_t)r * rs + cu * 8);
      const unsigned short* pv = reinterpret_cast<const unsigned short*>(&vvv);
#pragma unroll
      for (int e = 0; e < 8; ++e) Vt[cu * 8 + e][r] = pv[e];   // transpose
    }
    async_wait();
    __syncthreads();

    // S strip [16 x 64] = Q @ K^T
    v8f S[4];
#pragma unroll
    for (int fn = 0; fn < 4; ++fn) {
      S[fn] = zero;
#pragma unroll
      for (int ks = 0; ks < 4; ++ks) {
        FragBF bK;
        int rr = fn * 16 + lrow;
        int kb = ks * 32 + lhi * 8;
        bK.q[0] = *reinterpret_cast<const uint4*>(&Ks[rr][kb]);
        bK.q[1] = *reinterpret_cast<const uint4*>(&Ks[rr][kb + 16]);
        S[fn] = __builtin_amdgcn_wmma_f32_16x16x32_bf16(
            false, aQ[ks].v, false, bK.v, (short)0, S[fn], false, false);
      }
    }
    // scale + key-pad mask + online softmax
    float P[4][8], mt[8];
#pragma unroll
    for (int r = 0; r < 8; ++r) mt[r] = -1e30f;
#pragma unroll
    for (int fn = 0; fn < 4; ++fn) {
      int key = b * NT + kt * 64 + fn * 16 + lrow;
      bool pad = (seq[key] == 0);
#pragma unroll
      for (int r = 0; r < 8; ++r) {
        float s = S[fn][r] * 0.08838834764831845f;   // 1/sqrt(128)
        if (pad) s = -1e30f;
        P[fn][r] = s;
        mt[r] = fmaxf(mt[r], s);
      }
    }
#pragma unroll
    for (int r = 0; r < 8; ++r) {        // row max across the 16 N-lanes
      float v = mt[r];
      v = fmaxf(v, __shfl_xor(v, 1));
      v = fmaxf(v, __shfl_xor(v, 2));
      v = fmaxf(v, __shfl_xor(v, 4));
      v = fmaxf(v, __shfl_xor(v, 8));
      mt[r] = v;
    }
    float alpha[8];
#pragma unroll
    for (int r = 0; r < 8; ++r) {
      float mn = fmaxf(mrow[r], mt[r]);
      alpha[r] = __expf(mrow[r] - mn);
      mrow[r] = mn;
    }
    float st[8];
#pragma unroll
    for (int r = 0; r < 8; ++r) st[r] = 0.f;
#pragma unroll
    for (int fn = 0; fn < 4; ++fn)
#pragma unroll
      for (int r = 0; r < 8; ++r) {
        float p = __expf(P[fn][r] - mrow[r]);
        P[fn][r] = p;
        st[r] += p;
      }
#pragma unroll
    for (int r = 0; r < 8; ++r) {
      float v = st[r];
      v += __shfl_xor(v, 1);
      v += __shfl_xor(v, 2);
      v += __shfl_xor(v, 4);
      v += __shfl_xor(v, 8);
      lsum[r] = lsum[r] * alpha[r] + v;
    }
#pragma unroll
    for (int fn = 0; fn < 8; ++fn)
#pragma unroll
      for (int r = 0; r < 8; ++r) Oacc[fn][r] *= alpha[r];
    // P: C-frag layout -> LDS -> A-frag layout
#pragma unroll
    for (int fn = 0; fn < 4; ++fn)
#pragma unroll
      for (int r = 0; r < 8; ++r)
        Ps[qr + r + lhi * 8][fn * 16 + lrow] = f2bf(P[fn][r]);
    __syncthreads();
    // O += P @ V
#pragma unroll
    for (int ks = 0; ks < 2; ++ks) {
      FragBF aP;
      int rr = qr + lrow;
      int kb = ks * 32 + lhi * 8;
      aP.q[0] = *reinterpret_cast<const uint4*>(&Ps[rr][kb]);
      aP.q[1] = *reinterpret_cast<const uint4*>(&Ps[rr][kb + 16]);
#pragma unroll
      for (int fn = 0; fn < 8; ++fn) {
        FragBF bV;
        int dk  = fn * 16 + lrow;
        int kb2 = ks * 32 + lhi * 8;
        bV.q[0] = *reinterpret_cast<const uint4*>(&Vt[dk][kb2]);
        bV.q[1] = *reinterpret_cast<const uint4*>(&Vt[dk][kb2 + 16]);
        Oacc[fn] = __builtin_amdgcn_wmma_f32_16x16x32_bf16(
            false, aP.v, false, bV.v, (short)0, Oacc[fn], false, false);
      }
    }
  }
  // epilogue: O / l -> bf16 o[b,t, hd*128 + dk]
#pragma unroll
  for (int fn = 0; fn < 8; ++fn)
#pragma unroll
    for (int r = 0; r < 8; ++r) {
      int qq = qbase + qr + r + lhi * 8;
      int dk = fn * 16 + lrow;
      float li = lsum[r];
      float v = (li > 0.f) ? Oacc[fn][r] / li : 0.f;
      o[(size_t)(b * NT + qq) * ND + hd * 128 + dk] = f2bf(v);
    }
}

// ---------------------------------------------------------------------------
// Host launcher
// ---------------------------------------------------------------------------
extern "C" void kernel_launch(void* const* d_in, const int* in_sizes, int n_in,
                              void* d_out, int out_size, void* d_ws, size_t ws_size,
                              hipStream_t stream) {
  const int*   src_seq  = (const int*)d_in[0];
  const int*   src_pos  = (const int*)d_in[1];
  const float* word_emb = (const float*)d_in[2];
  const float* pos_emb  = (const float*)d_in[3];
  const float* qkv_w    = (const float*)d_in[4];
  const float* qkv_b    = (const float*)d_in[5];
  const float* fc_w     = (const float*)d_in[6];
  const float* fc_b     = (const float*)d_in[7];
  const float* ln1_g    = (const float*)d_in[8];
  const float* ln1_b    = (const float*)d_in[9];
  const float* conv1_w  = (const float*)d_in[10];
  const float* conv1_b  = (const float*)d_in[11];
  const float* conv2_w  = (const float*)d_in[12];
  const float* conv2_b  = (const float*)d_in[13];
  const float* ln2_g    = (const float*)d_in[14];
  const float* ln2_b    = (const float*)d_in[15];

  float* x       = (float*)d_out;                 // [32768, 256] in-place
  float* non_pad = x + (size_t)NBT * ND;          // [32768]

  // workspace layout (bytes; ~157.3 MB total, all 256B-aligned)
  char* ws = (char*)d_ws;
  unsigned short* h    = (unsigned short*)(ws);                         // 16.8 MB
  unsigned short* qkvb = (unsigned short*)(ws + 16777216ull);           // 50.3 MB
  unsigned short* hid  = (unsigned short*)(ws + 67108864ull);           // 67.1 MB
  unsigned short* wqkv = (unsigned short*)(ws + 134217728ull);          // 1.6 MB
  unsigned short* wfc  = (unsigned short*)(ws + 135790592ull);          // 0.5 MB
  unsigned short* wc1  = (unsigned short*)(ws + 136314880ull);          // 18.9 MB
  unsigned short* wc2  = (unsigned short*)(ws + 155189248ull);          // 2.1 MB

  // ---- weight conversion to bf16 ----
  {
    int n;
    n = NLAY * 768 * ND;        cvt_k<<<(n + 255) / 256, 256, 0, stream>>>(qkv_w,  wqkv, n);
    n = NLAY * ND * ND;         cvt_k<<<(n + 255) / 256, 256, 0, stream>>>(fc_w,   wfc,  n);
    n = NLAY * ND * NHID;       cvt_k<<<(n + 255) / 256, 256, 0, stream>>>(conv2_w, wc2, n);
    n = NLAY * NHID * ND * 9;   cvt_conv1_k<<<(n + 255) / 256, 256, 0, stream>>>(conv1_w, wc1, n);
  }

  // ---- embedding ----
  embed_k<<<NBT, 256, 0, stream>>>(src_seq, src_pos, word_emb, pos_emb, x, non_pad);

  for (int l = 0; l < NLAY; ++l) {
    // LN1 -> h (bf16)
    ln_k<<<NBT / 8, 256, 0, stream>>>(x, ln1_g + l * ND, ln1_b + l * ND, h);
    // QKV: [32768,256] @ [768,256]^T -> qkvb bf16 [32768,768]
    gemm_k<0><<<dim3(768 / 128, NBT / 128), 256, 0, stream>>>(
        h, ND, wqkv + (size_t)l * 768 * ND, ND, qkv_b + l * 768,
        nullptr, qkvb, 768, nullptr, nullptr, ND, /*mode*/0);
    // attention -> h (bf16, reused as attn-out buffer)
    attn_k<<<dim3(NT / 64, 2, NB), 128, 0, stream>>>(qkvb, src_seq, h);
    // out-proj + bias + residual + mask -> x (f32)
    gemm_k<0><<<dim3(ND / 128, NBT / 128), 256, 0, stream>>>(
        h, ND, wfc + (size_t)l * ND * ND, ND, fc_b + l * ND,
        x, nullptr, ND, x, non_pad, ND, /*mode*/2);
    // LN2 -> h (bf16)
    ln_k<<<NBT / 8, 256, 0, stream>>>(x, ln2_g + l * ND, ln2_b + l * ND, h);
    // conv1 (im2col GEMM, K=2304) + bias + relu -> hid bf16 [32768,1024]
    gemm_k<1><<<dim3(NHID / 128, NBT / 128), 256, 0, stream>>>(
        h, ND, wc1 + (size_t)l * NHID * 9 * ND, 9 * ND, conv1_b + l * NHID,
        nullptr, hid, NHID, nullptr, nullptr, 9 * ND, /*mode*/1);
    // conv2 (K=1024) + bias + residual + mask -> x (f32)
    gemm_k<0><<<dim3(ND / 128, NBT / 128), 256, 0, stream>>>(
        hid, NHID, wc2 + (size_t)l * ND * NHID, NHID, conv2_b + l * ND,
        x, nullptr, ND, x, non_pad, ND, /*mode*/2);
  }
  (void)in_sizes; (void)n_in; (void)out_size; (void)ws_size;
}
